// PhpNetGraphTokensCombine_65395172049661
// MI455X (gfx1250) — matrix operand order repaired and checked by reading, not compiled, not run on hardware
//
#include <hip/hip_runtime.h>

// ---------------------------------------------------------------------------
// MI455X (gfx1250): bf16 WMMA GEMMs (v_wmma_f32_16x16x32_bf16), f32 accumulate.
// Weights stay L2-resident (192MB); GEMM waves use 2x2 register tiling so each
// k-step does 4 WMMAs on 8 global_load_b128 (2 loads/WMMA).
// ---------------------------------------------------------------------------

typedef __bf16 bf16_t;
typedef __attribute__((ext_vector_type(16))) __bf16 v16bf;
typedef __attribute__((ext_vector_type(8)))  __bf16 v8bf;
typedef __attribute__((ext_vector_type(8)))  float  v8f;

#define N_NODES 2000
#define E_EDGES 16000
#define BATCH   16
#define SEQL    256
#define HDIM    2000
#define KP_H    2016   // HDIM padded to mult of 32
#define FIN     100
#define GHID    200
#define GATE3   600    // 3*GHID
#define KX      416    // 2*GHID padded to mult of 32
#define KH      224    // GHID padded to mult of 32
#define G3H     6000   // 3*HDIM
#define NSTEPS  3

__device__ __forceinline__ bf16_t f2bf(float f) {
  union { float f; unsigned u; } x; x.f = f;
  unsigned r = x.u + 0x7FFFu + ((x.u >> 16) & 1u);   // round-to-nearest-even
  unsigned short s = (unsigned short)(r >> 16);
  bf16_t o; __builtin_memcpy(&o, &s, 2); return o;
}
__device__ __forceinline__ float bf2f(bf16_t b) {
  unsigned short s; __builtin_memcpy(&s, &b, 2);
  union { unsigned u; float f; } x; x.u = ((unsigned)s) << 16; return x.f;
}
__device__ __forceinline__ float sigm(float x) { return 1.f / (1.f + expf(-x)); }

// A-fragment (16x32 bf16): lane = row; elems 0..15 at K=half*16+0..15
// (contiguous, matching the B fragment).
__device__ __forceinline__ v16bf load_afrag(const bf16_t* __restrict__ row, int k0, int half) {
  v8bf a0 = *(const v8bf*)(row + k0 + half * 16);
  v8bf a1 = *(const v8bf*)(row + k0 + half * 16 + 8);
  v16bf av;
#pragma unroll
  for (int i = 0; i < 8; ++i) { av[i] = a0[i]; av[8 + i] = a1[i]; }
  return av;
}

// ---------------- elementwise helpers ----------------

__global__ void castpad_kernel(const float* __restrict__ src, bf16_t* __restrict__ dst,
                               int R, int C, int Cp) {
  long long idx = (long long)blockIdx.x * 256 + threadIdx.x;
  if (idx >= (long long)R * Cp) return;
  int r = (int)(idx / Cp), c = (int)(idx % Cp);
  float v = (c < C) ? src[(long long)r * C + c] : 0.f;
  dst[idx] = f2bf(v);
}

__global__ void fill_f32_kernel(float* p, long long n) {
  long long idx = (long long)blockIdx.x * 256 + threadIdx.x;
  if (idx < n) p[idx] = 0.f;
}
__global__ void fill_bf16_kernel(bf16_t* p, long long n) {
  long long idx = (long long)blockIdx.x * 256 + threadIdx.x;
  if (idx < n) p[idx] = f2bf(0.f);
}

// h = [feats | 0]  (f32 and padded bf16 copies)
__global__ void init_h_kernel(const float* __restrict__ feats,
                              float* __restrict__ h, bf16_t* __restrict__ h_bf) {
  long long idx = (long long)blockIdx.x * 256 + threadIdx.x;
  if (idx >= (long long)N_NODES * KP_H) return;
  int n = (int)(idx / KP_H), j = (int)(idx % KP_H);
  float v = (j < FIN) ? feats[(long long)n * FIN + j] : 0.f;
  if (j < HDIM) h[(long long)n * HDIM + j] = v;
  h_bf[idx] = f2bf(v);
}

// xs0[l,b,k] = embed_w[tokens[b,l], k]  (zero-padded to KX)
__global__ void embed_kernel(const int* __restrict__ tokens,
                             const float* __restrict__ embed_w,
                             bf16_t* __restrict__ xs) {
  long long idx = (long long)blockIdx.x * 256 + threadIdx.x;
  if (idx >= (long long)SEQL * BATCH * KX) return;
  int l = (int)(idx / (BATCH * KX));
  int r = (int)(idx % (BATCH * KX));
  int b = r / KX, k = r % KX;
  int tok = tokens[b * SEQL + l];
  float v = (k < FIN) ? embed_w[(long long)tok * FIN + k] : 0.f;
  xs[idx] = f2bf(v);
}

// a[dst[e]] += t_{etype[e]}[src[e]]
__global__ void scatter_kernel(const float* __restrict__ t0, const float* __restrict__ t1,
                               const int* __restrict__ src, const int* __restrict__ dst,
                               const int* __restrict__ etype, float* __restrict__ a) {
  int e = blockIdx.x;
  int j = blockIdx.y * 256 + threadIdx.x;
  if (j >= HDIM) return;
  const float* t = (etype[e] == 0) ? t0 : t1;
  float v = t[(long long)src[e] * HDIM + j];
  atomicAdd(&a[(long long)dst[e] * HDIM + j], v);
}

// GGNN GRU cell: h = (1-z)*n + z*h, writes f32 + padded bf16 copy
__global__ void ggnn_cell_kernel(const float* __restrict__ gi, const float* __restrict__ gh,
                                 float* __restrict__ h, bf16_t* __restrict__ h_bf) {
  long long idx = (long long)blockIdx.x * 256 + threadIdx.x;
  if (idx >= (long long)N_NODES * HDIM) return;
  int n = (int)(idx / HDIM), j = (int)(idx % HDIM);
  const float* gin = gi + (long long)n * G3H;
  const float* ghn = gh + (long long)n * G3H;
  float r  = sigm(gin[j] + ghn[j]);
  float z  = sigm(gin[HDIM + j] + ghn[HDIM + j]);
  float nn = tanhf(gin[2 * HDIM + j] + r * ghn[2 * HDIM + j]);
  float hv = (1.f - z) * nn + z * h[idx];
  h[idx] = hv;
  h_bf[(long long)n * KP_H + j] = f2bf(hv);
}

// segment boundaries for sorted batch ids
__global__ void seg_bounds_kernel(const int* __restrict__ batch, int* __restrict__ seg) {
  int b = threadIdx.x;
  if (b >= BATCH) return;
  int lo = 0, hi = N_NODES;
  while (lo < hi) { int m = (lo + hi) >> 1; if (batch[m] < b) lo = m + 1; else hi = m; }
  seg[b] = lo;
  lo = 0; hi = N_NODES;
  while (lo < hi) { int m = (lo + hi) >> 1; if (batch[m] < b + 1) lo = m + 1; else hi = m; }
  seg[BATCH + b] = lo;
}

__global__ void seg_max_kernel(const float* __restrict__ h, const int* __restrict__ seg,
                               float* __restrict__ xg) {
  int b = blockIdx.x;
  int col = blockIdx.y * 256 + threadIdx.x;
  if (col >= HDIM) return;
  int s = seg[b], e = seg[BATCH + b];
  float m = -3.0e38f;
  for (int n = s; n < e; ++n) {
    float v = h[(long long)n * HDIM + col];
    m = v > m ? v : m;
  }
  xg[(long long)b * HDIM + col] = m;
}

// xcat(bf16) = [xg | hid transposed to (b, gate*200+u)]
__global__ void xcat_kernel(const float* __restrict__ xg, const float* __restrict__ hid,
                            bf16_t* __restrict__ xcat) {
  int idx = blockIdx.x * 256 + threadIdx.x;
  if (idx >= BATCH * 3200) return;
  int b = idx / 3200, j = idx % 3200;
  float v;
  if (j < HDIM) v = xg[b * HDIM + j];
  else {
    int g = (j - HDIM) / GHID, u = (j - HDIM) % GHID;
    v = hid[(long long)g * BATCH * GHID + b * GHID + u];
  }
  xcat[idx] = f2bf(v);
}

// ---------------- WMMA GEMM: C[M,N] = A[M,Kp](bf16) * B[N,Kp](bf16)^T + bias ----------------
// 256 threads = 8 waves arranged 2(M) x 4(N); each wave owns a 32x32 patch
// (2x2 WMMA tiles) so A/B fragments are each reused twice: 4 WMMA / 8 b128.
__global__ __launch_bounds__(256) void gemm_bf16_kernel(
    const bf16_t* __restrict__ A, const bf16_t* __restrict__ B,
    const float* __restrict__ bias, float* __restrict__ C,
    int M, int N, int Kp, int ldc, int relu) {
  const int lane = threadIdx.x & 31;
  const int wave = threadIdx.x >> 5;
  const int wm = wave & 1;        // 2 wave-rows
  const int wn = wave >> 1;       // 4 wave-cols
  const int half = lane >> 4;
  const int l15 = lane & 15;
  const int row0 = blockIdx.y * 64 + wm * 32;
  const int col0 = blockIdx.x * 128 + wn * 32;

  // clamp loads: OOB rows/cols only feed outputs that we never store
  int mA0 = row0 + l15;      if (mA0 > M - 1) mA0 = M - 1;
  int mA1 = row0 + 16 + l15; if (mA1 > M - 1) mA1 = M - 1;
  int nB0 = col0 + l15;      if (nB0 > N - 1) nB0 = N - 1;
  int nB1 = col0 + 16 + l15; if (nB1 > N - 1) nB1 = N - 1;
  const bf16_t* arow0 = A + (long long)mA0 * Kp;
  const bf16_t* arow1 = A + (long long)mA1 * Kp;
  const bf16_t* brow0 = B + (long long)nB0 * Kp;
  const bf16_t* brow1 = B + (long long)nB1 * Kp;
  __builtin_prefetch((const void*)brow0, 0, 2);   // global_prefetch_b8
  __builtin_prefetch((const void*)brow1, 0, 2);

  v8f acc00 = {}, acc01 = {}, acc10 = {}, acc11 = {};
  for (int k0 = 0; k0 < Kp; k0 += 32) {
    v16bf a0 = load_afrag(arow0, k0, half);
    v16bf a1 = load_afrag(arow1, k0, half);
    // B 32x16 frag: lane = col; 16 contiguous K at half*16
    v16bf b0 = *(const v16bf*)(brow0 + k0 + half * 16);
    v16bf b1 = *(const v16bf*)(brow1 + k0 + half * 16);
    acc00 = __builtin_amdgcn_wmma_f32_16x16x32_bf16(false, a0, false, b0, (short)0, acc00, false, false);
    acc01 = __builtin_amdgcn_wmma_f32_16x16x32_bf16(false, a0, false, b1, (short)0, acc01, false, false);
    acc10 = __builtin_amdgcn_wmma_f32_16x16x32_bf16(false, a1, false, b0, (short)0, acc10, false, false);
    acc11 = __builtin_amdgcn_wmma_f32_16x16x32_bf16(false, a1, false, b1, (short)0, acc11, false, false);
  }

  // C layout: VGPR i, lane l -> (M = i + 8*half, N = l15)
  auto store_tile = [&](const v8f& acc, int r0, int c0) {
    int cn = c0 + l15;
    if (cn < N) {
      float bv = bias ? bias[cn] : 0.f;
#pragma unroll
      for (int i = 0; i < 8; ++i) {
        int cm = r0 + half * 8 + i;
        if (cm < M) {
          float v = acc[i] + bv;
          if (relu) v = v > 0.f ? v : 0.f;
          C[(long long)cm * ldc + cn] = v;
        }
      }
    }
  };
  store_tile(acc00, row0,      col0);
  store_tile(acc01, row0,      col0 + 16);
  store_tile(acc10, row0 + 16, col0);
  store_tile(acc11, row0 + 16, col0 + 16);
}

// ---------------- persistent bidirectional GRU layer ----------------
// grid.x = 2 (direction), 256 threads (8 waves). State in LDS; 256 serial steps.
// N-tiles processed in pairs so the A fragment (x row / h row) is reused twice.
__global__ __launch_bounds__(256) void gru_layer_kernel(
    const bf16_t* __restrict__ xs,        // [L, BATCH, KX]
    bf16_t* __restrict__ xs_next,         // [L, BATCH, KX] (writes cols dir*GHID..+GHID)
    const bf16_t* __restrict__ Wih,       // [2, GATE3, KX]
    const bf16_t* __restrict__ Whh,       // [2, GATE3, KH]
    const float* __restrict__ bih,        // [2, GATE3]
    const float* __restrict__ bhh,        // [2, GATE3]
    float* __restrict__ hid_out) {        // [2, BATCH, GHID]
  __shared__ bf16_t h_bf[BATCH * KH];     //  7.0 KB (K-padded h for WMMA A)
  __shared__ float  h_f [BATCH * GHID];   // 12.8 KB
  __shared__ bf16_t gi_s[BATCH * GATE3];  // 19.2 KB
  __shared__ bf16_t gh_s[BATCH * GATE3];  // 19.2 KB   (total ~58 KB < 64 KB)

  const int tid  = threadIdx.x;
  const int dir  = blockIdx.x;
  const int lane = tid & 31;
  const int wave = tid >> 5;
  const int half = lane >> 4;
  const int l15  = lane & 15;

  const bf16_t* Wih_d = Wih + (long long)dir * GATE3 * KX;
  const bf16_t* Whh_d = Whh + (long long)dir * GATE3 * KH;
  const float*  bih_d = bih + dir * GATE3;
  const float*  bhh_d = bhh + dir * GATE3;

  for (int i = tid; i < BATCH * KH; i += 256) h_bf[i] = f2bf(0.f);
  for (int i = tid; i < BATCH * GHID; i += 256) h_f[i] = 0.f;
  __syncthreads();

  for (int t = 0; t < SEQL; ++t) {
    const int tt = dir ? (SEQL - 1 - t) : t;
    const bf16_t* x = xs + (long long)tt * BATCH * KX;

    for (int tp = wave; tp < 19; tp += 8) {         // 19 tile-pairs cover 38 tiles (GATE3=600)
      const int tile0 = tp * 2, tile1 = tp * 2 + 1;
      int nB0 = tile0 * 16 + l15; if (nB0 > GATE3 - 1) nB0 = GATE3 - 1;
      int nB1 = tile1 * 16 + l15; if (nB1 > GATE3 - 1) nB1 = GATE3 - 1;

      // gi = x @ Wih^T  (K = KX): 1 A-frag reused for 2 B tiles
      const bf16_t* arow  = x + l15 * KX;
      const bf16_t* brow0 = Wih_d + (long long)nB0 * KX;
      const bf16_t* brow1 = Wih_d + (long long)nB1 * KX;
      v8f acc0 = {}, acc1 = {};
      for (int k0 = 0; k0 < KX; k0 += 32) {
        v16bf av = load_afrag(arow, k0, half);
        v16bf b0 = *(const v16bf*)(brow0 + k0 + half * 16);
        v16bf b1 = *(const v16bf*)(brow1 + k0 + half * 16);
        acc0 = __builtin_amdgcn_wmma_f32_16x16x32_bf16(false, av, false, b0, (short)0, acc0, false, false);
        acc1 = __builtin_amdgcn_wmma_f32_16x16x32_bf16(false, av, false, b1, (short)0, acc1, false, false);
      }
      // gh = h @ Whh^T  (K = KH), A operand from LDS (ds_load_b128)
      const bf16_t* brow2 = Whh_d + (long long)nB0 * KH;
      const bf16_t* brow3 = Whh_d + (long long)nB1 * KH;
      v8f ach0 = {}, ach1 = {};
      for (int k0 = 0; k0 < KH; k0 += 32) {
        v8bf ha0 = *(const v8bf*)(h_bf + l15 * KH + k0 + half * 16);
        v8bf ha1 = *(const v8bf*)(h_bf + l15 * KH + k0 + half * 16 + 8);
        v16bf av;
#pragma unroll
        for (int i = 0; i < 8; ++i) { av[i] = ha0[i]; av[8 + i] = ha1[i]; }
        v16bf b0 = *(const v16bf*)(brow2 + k0 + half * 16);
        v16bf b1 = *(const v16bf*)(brow3 + k0 + half * 16);
        ach0 = __builtin_amdgcn_wmma_f32_16x16x32_bf16(false, av, false, b0, (short)0, ach0, false, false);
        ach1 = __builtin_amdgcn_wmma_f32_16x16x32_bf16(false, av, false, b1, (short)0, ach1, false, false);
      }
      // write both tiles of gi/gh (bf16) to LDS
#pragma unroll
      for (int tsel = 0; tsel < 2; ++tsel) {
        int cn = (tsel ? tile1 : tile0) * 16 + l15;
        const v8f& ai = tsel ? acc1 : acc0;
        const v8f& ah = tsel ? ach1 : ach0;
        if (cn < GATE3) {
          float bi = bih_d[cn], bh = bhh_d[cn];
#pragma unroll
          for (int i = 0; i < 8; ++i) {
            int cm = half * 8 + i;                   // batch row 0..15
            gi_s[cm * GATE3 + cn] = f2bf(ai[i] + bi);
            gh_s[cm * GATE3 + cn] = f2bf(ah[i] + bh);
          }
        }
      }
    }
    __syncthreads();

    for (int i = tid; i < BATCH * GHID; i += 256) {
      int b = i / GHID, u = i % GHID;
      const bf16_t* gi_b = gi_s + b * GATE3;
      const bf16_t* gh_b = gh_s + b * GATE3;
      float r  = sigm(bf2f(gi_b[u]) + bf2f(gh_b[u]));
      float z  = sigm(bf2f(gi_b[GHID + u]) + bf2f(gh_b[GHID + u]));
      float nn = tanhf(bf2f(gi_b[2 * GHID + u]) + r * bf2f(gh_b[2 * GHID + u]));
      float hv = (1.f - z) * nn + z * h_f[i];
      h_f[i] = hv;
      bf16_t hb = f2bf(hv);
      h_bf[b * KH + u] = hb;
      xs_next[(long long)tt * BATCH * KX + b * KX + dir * GHID + u] = hb;
    }
    __syncthreads();
  }
  for (int i = tid; i < BATCH * GHID; i += 256)
    hid_out[(long long)dir * BATCH * GHID + i] = h_f[i];
}

// ---------------------------------------------------------------------------

extern "C" void kernel_launch(void* const* d_in, const int* in_sizes, int n_in,
                              void* d_out, int out_size, void* d_ws, size_t ws_size,
                              hipStream_t stream) {
  (void)in_sizes; (void)n_in; (void)out_size; (void)ws_size;
  const float* feats    = (const float*)d_in[0];
  const int*   tokens   = (const int*)d_in[1];
  const int*   src      = (const int*)d_in[2];
  const int*   dst      = (const int*)d_in[3];
  const int*   etype    = (const int*)d_in[4];
  const int*   batch    = (const int*)d_in[5];
  const float* embed_w  = (const float*)d_in[6];
  const float* ggnn_W   = (const float*)d_in[7];
  const float* ggnn_b   = (const float*)d_in[8];
  const float* ggnn_Wih = (const float*)d_in[9];
  const float* ggnn_Whh = (const float*)d_in[10];
  const float* ggnn_bih = (const float*)d_in[11];
  const float* ggnn_bhh = (const float*)d_in[12];
  const float* gru_Wih  = (const float*)d_in[13];
  const float* gru_Whh  = (const float*)d_in[14];
  const float* gru_bih  = (const float*)d_in[15];
  const float* gru_bhh  = (const float*)d_in[16];
  const float* lin1_W   = (const float*)d_in[17];
  const float* lin1_b   = (const float*)d_in[18];
  const float* lin11_W  = (const float*)d_in[19];
  const float* lin11_b  = (const float*)d_in[20];
  const float* lin2_W   = (const float*)d_in[21];
  const float* lin2_b   = (const float*)d_in[22];
  float* out = (float*)d_out;

  char* wp = (char*)d_ws;
  auto carve = [&](size_t bytes) -> char* {
    char* p = wp;
    wp += (bytes + 255) & ~(size_t)255;
    return p;
  };

  float*  h      = (float*) carve((size_t)N_NODES * HDIM * 4);
  float*  a      = (float*) carve((size_t)N_NODES * HDIM * 4);
  float*  t0     = (float*) carve((size_t)N_NODES * HDIM * 4);
  float*  t1     = (float*) carve((size_t)N_NODES * HDIM * 4);
  float*  gi     = (float*) carve((size_t)N_NODES * G3H * 4);
  float*  gh     = (float*) carve((size_t)N_NODES * G3H * 4);
  bf16_t* h_bf   = (bf16_t*)carve((size_t)N_NODES * KP_H * 2);
  bf16_t* a_bf   = (bf16_t*)carve((size_t)N_NODES * KP_H * 2);
  bf16_t* Wg_bf  = (bf16_t*)carve((size_t)2 * HDIM * KP_H * 2);
  bf16_t* Wih_bf = (bf16_t*)carve((size_t)G3H * KP_H * 2);
  bf16_t* Whh_bf = (bf16_t*)carve((size_t)G3H * KP_H * 2);
  float*  xg     = (float*) carve((size_t)BATCH * HDIM * 4);
  int*    seg    = (int*)   carve((size_t)2 * BATCH * 4);
  bf16_t* gWih   = (bf16_t*)carve((size_t)6 * GATE3 * KX * 2);
  bf16_t* gWhh   = (bf16_t*)carve((size_t)6 * GATE3 * KH * 2);
  bf16_t* xsA    = (bf16_t*)carve((size_t)SEQL * BATCH * KX * 2);
  bf16_t* xsB    = (bf16_t*)carve((size_t)SEQL * BATCH * KX * 2);
  float*  hid    = (float*) carve((size_t)6 * BATCH * GHID * 4);
  bf16_t* xcat   = (bf16_t*)carve((size_t)BATCH * 3200 * 2);
  bf16_t* l1w    = (bf16_t*)carve((size_t)1000 * 3200 * 2);
  bf16_t* l11w   = (bf16_t*)carve((size_t)500 * 1024 * 2);
  bf16_t* l2w    = (bf16_t*)carve((size_t)2 * 512 * 2);
  float*  y1     = (float*) carve((size_t)BATCH * 1000 * 4);
  bf16_t* y1b    = (bf16_t*)carve((size_t)BATCH * 1024 * 2);
  float*  y2     = (float*) carve((size_t)BATCH * 500 * 4);
  bf16_t* y2b    = (bf16_t*)carve((size_t)BATCH * 512 * 2);

  auto nblk = [](long long n) { return (unsigned)((n + 255) / 256); };

  // ---- weight conversions to K-padded bf16 (L2-resident thereafter) ----
  castpad_kernel<<<nblk((long long)2 * HDIM * KP_H), 256, 0, stream>>>(ggnn_W,  Wg_bf,  2 * HDIM, HDIM, KP_H);
  castpad_kernel<<<nblk((long long)G3H * KP_H),      256, 0, stream>>>(ggnn_Wih, Wih_bf, G3H, HDIM, KP_H);
  castpad_kernel<<<nblk((long long)G3H * KP_H),      256, 0, stream>>>(ggnn_Whh, Whh_bf, G3H, HDIM, KP_H);
  castpad_kernel<<<nblk((long long)6 * GATE3 * KX),  256, 0, stream>>>(gru_Wih, gWih, 6 * GATE3, 2 * GHID, KX);
  castpad_kernel<<<nblk((long long)6 * GATE3 * KH),  256, 0, stream>>>(gru_Whh, gWhh, 6 * GATE3, GHID, KH);
  castpad_kernel<<<nblk((long long)1000 * 3200),     256, 0, stream>>>(lin1_W,  l1w,  1000, 3200, 3200);
  castpad_kernel<<<nblk((long long)500 * 1024),      256, 0, stream>>>(lin11_W, l11w, 500, 1000, 1024);
  castpad_kernel<<<nblk((long long)2 * 512),         256, 0, stream>>>(lin2_W,  l2w,  2, 500, 512);

  // ---- activations init ----
  init_h_kernel<<<nblk((long long)N_NODES * KP_H), 256, 0, stream>>>(feats, h, h_bf);
  embed_kernel<<<nblk((long long)SEQL * BATCH * KX), 256, 0, stream>>>(tokens, embed_w, xsA);
  fill_bf16_kernel<<<nblk((long long)SEQL * BATCH * KX), 256, 0, stream>>>(xsB, (long long)SEQL * BATCH * KX);

  // ---- GGNN: 3 steps ----
  dim3 grid_t((HDIM + 127) / 128, (N_NODES + 63) / 64);   // 16 x 32
  dim3 grid_g((G3H + 127) / 128,  (N_NODES + 63) / 64);   // 47 x 32
  for (int s = 0; s < NSTEPS; ++s) {
    gemm_bf16_kernel<<<grid_t, 256, 0, stream>>>(h_bf, Wg_bf, ggnn_b, t0,
                                                 N_NODES, HDIM, KP_H, HDIM, 0);
    gemm_bf16_kernel<<<grid_t, 256, 0, stream>>>(h_bf, Wg_bf + (size_t)HDIM * KP_H,
                                                 ggnn_b + HDIM, t1,
                                                 N_NODES, HDIM, KP_H, HDIM, 0);
    fill_f32_kernel<<<nblk((long long)N_NODES * HDIM), 256, 0, stream>>>(a, (long long)N_NODES * HDIM);
    scatter_kernel<<<dim3(E_EDGES, (HDIM + 255) / 256), 256, 0, stream>>>(t0, t1, src, dst, etype, a);
    castpad_kernel<<<nblk((long long)N_NODES * KP_H), 256, 0, stream>>>(a, a_bf, N_NODES, HDIM, KP_H);
    gemm_bf16_kernel<<<grid_g, 256, 0, stream>>>(a_bf, Wih_bf, ggnn_bih, gi,
                                                 N_NODES, G3H, KP_H, G3H, 0);
    gemm_bf16_kernel<<<grid_g, 256, 0, stream>>>(h_bf, Whh_bf, ggnn_bhh, gh,
                                                 N_NODES, G3H, KP_H, G3H, 0);
    ggnn_cell_kernel<<<nblk((long long)N_NODES * HDIM), 256, 0, stream>>>(gi, gh, h, h_bf);
  }
  seg_bounds_kernel<<<1, 32, 0, stream>>>(batch, seg);
  seg_max_kernel<<<dim3(BATCH, (HDIM + 255) / 256), 256, 0, stream>>>(h, seg, xg);

  // ---- token branch: 3 bidirectional GRU layers (ping-pong xsA/xsB) ----
  for (int l = 0; l < 3; ++l) {
    const bf16_t* xin = (l == 1) ? xsB : xsA;
    bf16_t* xout      = (l == 1) ? xsA : xsB;
    gru_layer_kernel<<<2, 256, 0, stream>>>(
        xin, xout,
        gWih + (size_t)l * 2 * GATE3 * KX,
        gWhh + (size_t)l * 2 * GATE3 * KH,
        gru_bih + l * 2 * GATE3,
        gru_bhh + l * 2 * GATE3,
        hid + (size_t)l * 2 * BATCH * GHID);
  }

  // ---- head ----
  xcat_kernel<<<nblk((long long)BATCH * 3200), 256, 0, stream>>>(xg, hid, xcat);
  gemm_bf16_kernel<<<dim3((1000 + 127) / 128, 1), 256, 0, stream>>>(xcat, l1w, lin1_b, y1,
                                                                    BATCH, 1000, 3200, 1000, 1);
  castpad_kernel<<<nblk((long long)BATCH * 1024), 256, 0, stream>>>(y1, y1b, BATCH, 1000, 1024);
  gemm_bf16_kernel<<<dim3((500 + 127) / 128, 1), 256, 0, stream>>>(y1b, l11w, lin11_b, y2,
                                                                   BATCH, 500, 1024, 500, 1);
  castpad_kernel<<<nblk((long long)BATCH * 512), 256, 0, stream>>>(y2, y2b, BATCH, 500, 512);
  gemm_bf16_kernel<<<dim3(1, 1), 256, 0, stream>>>(y2b, l2w, lin2_b, out,
                                                   BATCH, 2, 512, 2, 1);
}
